// SpikingParrot_12008728560086
// MI455X (gfx1250) — matrix operand order repaired and checked by reading, not compiled, not run on hardware
//
#include <hip/hip_runtime.h>
#include <cmath>

// ---------------------------------------------------------------------------
// SpikingParrot forward for MI455X (gfx1250, wave32, WMMA).
// Mixed precision: all linears are f16 WMMA GEMMs with f32 accumulate;
// weights cast to f16 once per call; DyT fused into GEMM epilogue.
// GEMM uses CDNA5 async global->LDS staging (ASYNCcnt) + double buffering.
// ---------------------------------------------------------------------------

typedef __attribute__((ext_vector_type(16))) _Float16 v16h;
typedef __attribute__((ext_vector_type(8)))  _Float16 v8h;
typedef __attribute__((ext_vector_type(8)))  float    v8f;

#define BN_   8
#define S_SRC 256
#define S_TGT 128
#define E_    512
#define H_    1024
#define V_    32000

#define TBM 64
#define TBN 128
#define TBK 32
#define LDP 40   // LDS row pitch in f16 (80B, multiple of 16B)

__device__ __forceinline__ float sigf(float x) { return 1.0f / (1.0f + __expf(-x)); }

// ------------------------- WMMA GEMM --------------------------------------
// C[M,N] (row stride ldc) = A[M,K]f16 * W[N,K]f16^T + bias[N]
// optional DyT epilogue: gamma[n]*tanh(alpha*val)+beta[n]
// optional f16 mirror C16 (row stride ldc16). Requires N%128==0, K%32==0.
__global__ __launch_bounds__(256) void gemm_wmma(
    const _Float16* __restrict__ A, const _Float16* __restrict__ W,
    const float* __restrict__ bias,
    float* __restrict__ C, int ldc,
    _Float16* __restrict__ C16, int ldc16,
    int M, int N, int K,
    const float* __restrict__ dalpha, const float* __restrict__ dgamma,
    const float* __restrict__ dbeta)
{
    __shared__ __align__(16) _Float16 As[2][TBM * LDP];
    __shared__ __align__(16) _Float16 Ws[2][TBN * LDP];

    const int t    = threadIdx.x;
    const int lane = t & 31;
    const int wv   = t >> 5;
    const int wy   = wv & 3;   // 16-row sub-tile
    const int wx   = wv >> 2;  // 64-col sub-tile
    const int m0   = blockIdx.y * TBM;
    const int n0   = blockIdx.x * TBN;

    v8f acc[4];
#pragma unroll
    for (int i = 0; i < 4; ++i)
#pragma unroll
        for (int j = 0; j < 8; ++j) acc[i][j] = 0.0f;

    const int ar = t >> 2;          // A stage row 0..63
    const int ac = (t & 3) * 8;     // A stage col chunk
    int arow = m0 + ar;             // clamp OOB rows (dup data never stored)
    if (arow >= M) arow = M - 1;

    // Async global->LDS staging of one K-chunk into buffer `buf`.
    auto stage = [&](int buf, int kb) {
        {
            unsigned lA = (unsigned)(uintptr_t)&As[buf][ar * LDP + ac];
            const _Float16* gA = A + (long)arow * K + kb + ac;
            asm volatile("global_load_async_to_lds_b128 %0, %1, off"
                         :: "v"(lA), "v"(gA) : "memory");
        }
#pragma unroll
        for (int j = 0; j < 2; ++j) {
            int idx = t + j * 256;
            int r = idx >> 2, c = (idx & 3) * 8;
            unsigned lW = (unsigned)(uintptr_t)&Ws[buf][r * LDP + c];
            const _Float16* gW = W + (long)(n0 + r) * K + kb + c;
            asm volatile("global_load_async_to_lds_b128 %0, %1, off"
                         :: "v"(lW), "v"(gW) : "memory");
        }
    };

    stage(0, 0);
    asm volatile("s_wait_asynccnt 0" ::: "memory");
    __syncthreads();

    int cur = 0;
    for (int kb = 0; kb < K; kb += TBK) {
        // prime the async engine for the next chunk while we compute
        if (kb + TBK < K) stage(cur ^ 1, kb + TBK);
        // L2 prefetch two stages ahead
        if (kb + 2 * TBK < K) {
            __builtin_prefetch(A + (long)arow * K + kb + 2 * TBK + ac, 0, 2);
            __builtin_prefetch(W + (long)(n0 + (t >> 1)) * K + kb + 2 * TBK, 0, 2);
        }

        // A fragment: lanes 0-15 hold K 0..15, lanes 16-31 hold K 16..31
        union Fr { v16h v; v8h h[2]; };
        Fr af, bf0, bf1, bf2, bf3;
        const _Float16* Ab = As[cur];
        const _Float16* Wb = Ws[cur];
        const int ma    = wy * 16 + (lane & 15);
        const int koffA = (lane >> 4) * 16;
        af.h[0] = *(const v8h*)(Ab + ma * LDP + koffA);
        af.h[1] = *(const v8h*)(Ab + ma * LDP + koffA + 8);
        // B fragments: lanes 0-15 hold K 0..15, lanes 16-31 hold K 16..31;
        // hoist all four loads so the WMMAs issue back-to-back.
        const int kbB   = (lane >> 4) * 16;
        const int nbase = wx * 64 + (lane & 15);
        bf0.h[0] = *(const v8h*)(Wb + (nbase +  0) * LDP + kbB);
        bf0.h[1] = *(const v8h*)(Wb + (nbase +  0) * LDP + kbB + 8);
        bf1.h[0] = *(const v8h*)(Wb + (nbase + 16) * LDP + kbB);
        bf1.h[1] = *(const v8h*)(Wb + (nbase + 16) * LDP + kbB + 8);
        bf2.h[0] = *(const v8h*)(Wb + (nbase + 32) * LDP + kbB);
        bf2.h[1] = *(const v8h*)(Wb + (nbase + 32) * LDP + kbB + 8);
        bf3.h[0] = *(const v8h*)(Wb + (nbase + 48) * LDP + kbB);
        bf3.h[1] = *(const v8h*)(Wb + (nbase + 48) * LDP + kbB + 8);

        acc[0] = __builtin_amdgcn_wmma_f32_16x16x32_f16(false, af.v, false, bf0.v,
                                                        (short)0, acc[0], false, false);
        acc[1] = __builtin_amdgcn_wmma_f32_16x16x32_f16(false, af.v, false, bf1.v,
                                                        (short)0, acc[1], false, false);
        acc[2] = __builtin_amdgcn_wmma_f32_16x16x32_f16(false, af.v, false, bf2.v,
                                                        (short)0, acc[2], false, false);
        acc[3] = __builtin_amdgcn_wmma_f32_16x16x32_f16(false, af.v, false, bf3.v,
                                                        (short)0, acc[3], false, false);

        // publish next buffer: own async loads done + workgroup barrier
        asm volatile("s_wait_asynccnt 0" ::: "memory");
        __syncthreads();
        cur ^= 1;
    }

    const float alpha = dgamma ? dalpha[0] : 0.0f;
#pragma unroll
    for (int s = 0; s < 4; ++s) {
        const int n = n0 + wx * 64 + s * 16 + (lane & 15);
        const float bv = bias ? bias[n] : 0.0f;
        const float ga = dgamma ? dgamma[n] : 0.0f;
        const float be = dgamma ? dbeta[n]  : 0.0f;
#pragma unroll
        for (int r = 0; r < 8; ++r) {
            const int m = m0 + wy * 16 + r + ((lane >> 4) << 3);
            if (m < M) {
                float val = acc[s][r] + bv;
                if (dgamma) val = ga * tanhf(alpha * val) + be;
                C[(long)m * ldc + n] = val;
                if (C16) C16[(long)m * ldc16 + n] = (_Float16)val;
            }
        }
    }
}

// ------------------------- small helper kernels ----------------------------
__global__ void cast_f32_f16_k(const float* __restrict__ in, _Float16* __restrict__ out, long n)
{
    for (long i = blockIdx.x * (long)blockDim.x + threadIdx.x; i < n;
         i += (long)gridDim.x * blockDim.x)
        out[i] = (_Float16)in[i];
}

__global__ void fill_f32_k(float* p, float v, long n)
{
    for (long i = blockIdx.x * (long)blockDim.x + threadIdx.x; i < n;
         i += (long)gridDim.x * blockDim.x) p[i] = v;
}

__global__ void fill_f16_k(_Float16* p, float v, long n)
{
    for (long i = blockIdx.x * (long)blockDim.x + threadIdx.x; i < n;
         i += (long)gridDim.x * blockDim.x) p[i] = (_Float16)v;
}

// embedding gather + DyT; out16 optional
__global__ void embed_dyt_k(const int* __restrict__ tok, const float* __restrict__ emb,
                            const float* __restrict__ alpha, const float* __restrict__ gamma,
                            const float* __restrict__ beta,
                            float* __restrict__ out, _Float16* __restrict__ out16,
                            long BS, int Ed)
{
    const float a = alpha[0];
    const long total = BS * Ed;
    for (long i = blockIdx.x * (long)blockDim.x + threadIdx.x; i < total;
         i += (long)gridDim.x * blockDim.x) {
        long r = i / Ed; int e = (int)(i % Ed);
        float x = emb[(long)tok[r] * Ed + e];
        float y = gamma[e] * tanhf(a * x) + beta[e];
        out[i] = y;
        if (out16) out16[i] = (_Float16)y;
    }
}

// out[b,s,:] = in[b,S-1-s,:], f32 + f16
__global__ void flip_seq_k(const float* __restrict__ in, float* __restrict__ out,
                           _Float16* __restrict__ out16, int nB, int S, int K)
{
    const long total = (long)nB * S * K;
    for (long i = blockIdx.x * (long)blockDim.x + threadIdx.x; i < total;
         i += (long)gridDim.x * blockDim.x) {
        int k = (int)(i % K); long bs = i / K; int s = (int)(bs % S); int b = (int)(bs / S);
        float v = in[((long)b * S + (S - 1 - s)) * K + k];
        out[i] = v;
        if (out16) out16[i] = (_Float16)v;
    }
}

// out16[b,s,:] = concat(A[b,s,:K1], Bsrc[b, flip? S-1-s : s, :K2])
__global__ void concat_seq_f16_k(const float* __restrict__ A, const float* __restrict__ Bs,
                                 _Float16* __restrict__ out, int nB, int S, int K1, int K2,
                                 int flipB)
{
    const long K = K1 + K2;
    const long total = (long)nB * S * K;
    for (long i = blockIdx.x * (long)blockDim.x + threadIdx.x; i < total;
         i += (long)gridDim.x * blockDim.x) {
        long k = i % K; long bs = i / K; int s = (int)(bs % S); int b = (int)(bs / S);
        float v;
        if (k < K1) v = A[((long)b * S + s) * K1 + k];
        else {
            int s2 = flipB ? (S - 1 - s) : s;
            v = Bs[((long)b * S + s2) * K2 + (k - K1)];
        }
        out[i] = (_Float16)v;
    }
}

// row-strided 2-way concat: out[m,:] = concat(A[m*lda : +K1], B[m*ldb : +K2]); f16 + opt f32
__global__ void concat2_rows_k(const float* __restrict__ A, long lda, int K1,
                               const float* __restrict__ Bs, long ldb, int K2,
                               _Float16* __restrict__ out16, float* __restrict__ out32, int M)
{
    const int K = K1 + K2;
    const long total = (long)M * K;
    for (long i = blockIdx.x * (long)blockDim.x + threadIdx.x; i < total;
         i += (long)gridDim.x * blockDim.x) {
        int k = (int)(i % K); int m = (int)(i / K);
        float v = (k < K1) ? A[(long)m * lda + k] : Bs[(long)m * ldb + (k - K1)];
        out16[i] = (_Float16)v;
        if (out32) out32[i] = v;
    }
}

// LSTM cell pointwise: z = zx + zh (biases already in GEMMs); gate order i,f,g,o.
// Updates syn(c), mem(h) in place; writes f16 mem mirror and strided spike f16.
__global__ void lstm_point_k(const float* __restrict__ zx, long zx_rs,
                             const float* __restrict__ zh,
                             float* __restrict__ syn, float* __restrict__ mem,
                             _Float16* __restrict__ mem16,
                             _Float16* __restrict__ spk16, long spk_rs,
                             const float* __restrict__ thr, int nB, int Hd)
{
    int i = blockIdx.x * blockDim.x + threadIdx.x;
    int total = nB * Hd;
    if (i >= total) return;
    int b = i / Hd, h = i % Hd;
    const float* zxr = zx + (long)b * zx_rs;
    const float* zhr = zh + (long)b * 4 * Hd;
    float zi = zxr[h]          + zhr[h];
    float zf = zxr[Hd + h]     + zhr[Hd + h];
    float zg = zxr[2 * Hd + h] + zhr[2 * Hd + h];
    float zo = zxr[3 * Hd + h] + zhr[3 * Hd + h];
    float c  = syn[i];
    float c2 = sigf(zf) * c + sigf(zi) * tanhf(zg);
    float h2 = sigf(zo) * tanhf(c2);
    syn[i] = c2;
    mem[i] = h2;
    mem16[i] = (_Float16)h2;
    float s = (h2 - thr[0]) > 0.0f ? 1.0f : 0.0f;
    spk16[(long)b * spk_rs + h] = (_Float16)s;
}

// fused attention step: scores -> softmax -> context -> DyT. One block per batch row.
__global__ __launch_bounds__(256) void attn_step_k(
    const float* __restrict__ q, const float* __restrict__ k, const float* __restrict__ v,
    const float* __restrict__ dalpha, const float* __restrict__ dgamma,
    const float* __restrict__ dbeta, float* __restrict__ ctx, int S, int Hd)
{
    __shared__ float sc[256];
    __shared__ float red[256];
    const int b = blockIdx.x, t = threadIdx.x;
    const float* qb = q + (long)b * Hd;
    float sco = -3.0e38f;
    if (t < S) {
        const float* kb = k + ((long)b * S + t) * Hd;
        float acc = 0.0f;
        for (int h = 0; h < Hd; ++h) acc += qb[h] * kb[h];
        sco = acc * rsqrtf((float)Hd);
    }
    red[t] = sco;
    __syncthreads();
    for (int o = 128; o > 0; o >>= 1) { if (t < o) red[t] = fmaxf(red[t], red[t + o]); __syncthreads(); }
    const float mx = red[0];
    __syncthreads();
    float ex = (t < S) ? __expf(sco - mx) : 0.0f;
    red[t] = ex;
    __syncthreads();
    for (int o = 128; o > 0; o >>= 1) { if (t < o) red[t] += red[t + o]; __syncthreads(); }
    const float inv = 1.0f / red[0];
    sc[t] = ex * inv;
    __syncthreads();
    const float alpha = dalpha[0];
    for (int h = t; h < Hd; h += 256) {
        float a2 = 0.0f;
        for (int s = 0; s < S; ++s) a2 += sc[s] * v[((long)b * S + s) * Hd + h];
        ctx[(long)b * Hd + h] = dgamma[h] * tanhf(alpha * a2) + dbeta[h];
    }
}

// ---------------------------------------------------------------------------
// Host orchestration
// ---------------------------------------------------------------------------
extern "C" void kernel_launch(void* const* d_in, const int* in_sizes, int n_in,
                              void* d_out, int out_size, void* d_ws, size_t ws_size,
                              hipStream_t stream)
{
    (void)in_sizes; (void)n_in; (void)out_size; (void)ws_size;
    const int* src = (const int*)d_in[0];
    const int* tgt = (const int*)d_in[1];
    auto F = [&](int i) { return (const float*)d_in[i]; };

    // Flattened param indices (sorted-key DFS of the params pytree).
    const int DEC_EMB = 2, DEC_NA = 3, DEC_NB = 4, DEC_NG = 5;
    const int ENC_EMB = 48, ENC_NA = 49, ENC_NB = 50, ENC_NG = 51;
    const int FC_W = 100, FC_B = 101, MEMFC_W = 102, MEMFC_B = 103;
    const int OUTFC_W = 104, OUTFC_B = 105, SYNFC_W = 106, SYNFC_B = 107;

    char* wp = (char*)d_ws;
    auto bump = [&](size_t bytes) -> void* {
        void* r = wp; wp += (bytes + 255) & ~(size_t)255; return r;
    };
    auto A32 = [&](size_t n) { return (float*)bump(n * sizeof(float)); };
    auto A16 = [&](size_t n) { return (_Float16*)bump(n * sizeof(_Float16)); };

    auto castW = [&](int idx, size_t n) -> const _Float16* {
        _Float16* d = A16(n);
        cast_f32_f16_k<<<dim3(1024), dim3(256), 0, stream>>>(F(idx), d, (long)n);
        return d;
    };
    auto gemm = [&](const _Float16* Aa, const _Float16* Ww, const float* bias,
                    float* Cc, int ldc, _Float16* Cc16, int ldc16,
                    int M, int N, int K,
                    const float* da, const float* dg, const float* db) {
        dim3 g(N / TBN, (M + TBM - 1) / TBM);
        gemm_wmma<<<g, dim3(256), 0, stream>>>(Aa, Ww, bias, Cc, ldc, Cc16, ldc16,
                                               M, N, K, da, dg, db);
    };

    const int BSs = BN_ * S_SRC;   // 2048
    const int BSt = BN_ * S_TGT;   // 1024

    // ---- global (layer-independent) f16 weights ----
    const _Float16* fc16    = castW(FC_W,    (size_t)V_ * H_);
    const _Float16* synfc16 = castW(SYNFC_W, (size_t)H_ * 2 * H_);
    const _Float16* memfc16 = castW(MEMFC_W, (size_t)H_ * 2 * H_);
    const _Float16* outfc16 = castW(OUTFC_W, (size_t)H_ * 2 * H_);

    // ---- activations / scratch ----
    float*    curF32 = A32((size_t)BSs * H_); _Float16* curF16 = A16((size_t)BSs * H_);
    float*    curB32 = A32((size_t)BSs * H_); _Float16* curB16 = A16((size_t)BSs * H_);
    float*    nxtF32 = A32((size_t)BSs * H_); _Float16* nxtF16 = A16((size_t)BSs * H_);
    float*    nxtB32 = A32((size_t)BSs * H_); _Float16* nxtB16 = A16((size_t)BSs * H_);
    float*    zx      = A32((size_t)BSs * 4 * H_);
    _Float16* spkseq  = A16((size_t)BSs * H_);
    float*    linspk  = A32((size_t)BSs * H_);
    _Float16* catbuf  = A16((size_t)BSs * 2 * H_);
    float*    encO32[2]; _Float16* encO16[2]; float* k32[2]; float* v32[2];
    for (int i = 0; i < 2; ++i) {
        encO32[i] = A32((size_t)BSs * H_); encO16[i] = A16((size_t)BSs * H_);
        k32[i] = A32((size_t)BSs * H_);    v32[i] = A32((size_t)BSs * H_);
    }
    float* synF = A32(BN_ * H_); float* memF = A32(BN_ * H_); _Float16* memF16 = A16(BN_ * H_);
    float* synB = A32(BN_ * H_); float* memB = A32(BN_ * H_); _Float16* memB16 = A16(BN_ * H_);
    _Float16* cat8 = A16(BN_ * 2 * H_);
    float* synD[2]; float* memD[2]; _Float16* memD16[2];
    for (int i = 0; i < 2; ++i) {
        synD[i] = A32(BN_ * H_); memD[i] = A32(BN_ * H_); memD16[i] = A16(BN_ * H_);
    }
    float*    zh   = A32(BN_ * 4 * H_);
    float*    zxd  = A32(BN_ * 4 * H_);
    float*    temb = A32((size_t)BSt * E_);
    float*    q32  = A32(BN_ * H_);
    float*    ctx32 = A32(BN_ * H_);
    _Float16* inp16 = A16(BN_ * 2 * H_);  float* inp32 = A32(BN_ * 2 * H_);
    _Float16* spkd  = A16(BN_ * H_);      float* linspkd = A32(BN_ * H_);
    _Float16* mixind = A16(BN_ * 3 * H_);
    float*    outL32 = A32(BN_ * H_);     _Float16* outL16 = A16(BN_ * H_);
    float*    dec32  = A32((size_t)BSt * H_); _Float16* dec16 = A16((size_t)BSt * H_);

    // =================== encoder ===================
    embed_dyt_k<<<dim3(512), dim3(256), 0, stream>>>(src, F(ENC_EMB), F(ENC_NA),
                                                     F(ENC_NG), F(ENC_NB),
                                                     curF32, curF16, (long)BSs, E_);
    flip_seq_k<<<dim3(512), dim3(256), 0, stream>>>(curF32, curB32, curB16, BN_, S_SRC, E_);

    for (int li = 0; li < 2; ++li) {
        const int ind = (li == 0) ? E_ : H_;
        for (int dir = 0; dir < 2; ++dir) {   // 0 = forward ('f'), 1 = backward ('b')
            const int base = 52 + li * 24 + (dir == 0 ? 12 : 0); // 'b' first, then 'f'
            const int iWhh = base + 0, iWih = base + 1, ibhh = base + 2, ibih = base + 3;
            const int ilinW = base + 4, ilinb = base + 5, imixW = base + 6, imixb = base + 7;
            const int ina = base + 8, inb = base + 9, ing = base + 10, ithr = base + 11;

            const _Float16* wih16 = castW(iWih,  (size_t)4 * H_ * ind);
            const _Float16* whh16 = castW(iWhh,  (size_t)4 * H_ * H_);
            const _Float16* lin16 = castW(ilinW, (size_t)H_ * H_);
            const _Float16* mix16 = castW(imixW, (size_t)H_ * (H_ + ind));

            float*    x32 = dir ? curB32 : curF32; _Float16* x16 = dir ? curB16 : curF16;
            float*    o32 = dir ? nxtB32 : nxtF32; _Float16* o16 = dir ? nxtB16 : nxtF16;
            float*    syn = dir ? synB : synF;     float* mem = dir ? memB : memF;
            _Float16* m16 = dir ? memB16 : memF16;

            // hoisted x @ Wih^T + bih over whole sequence
            gemm(x16, wih16, F(ibih), zx, 4 * H_, nullptr, 0,
                 BSs, 4 * H_, ind, nullptr, nullptr, nullptr);

            fill_f32_k<<<dim3(32), dim3(256), 0, stream>>>(syn, 0.0f, (long)BN_ * H_);
            fill_f32_k<<<dim3(32), dim3(256), 0, stream>>>(mem, 0.0f, (long)BN_ * H_);
            fill_f16_k<<<dim3(32), dim3(256), 0, stream>>>(m16, 0.0f, (long)BN_ * H_);

            for (int ts = 0; ts < S_SRC; ++ts) {
                gemm(m16, whh16, F(ibhh), zh, 4 * H_, nullptr, 0,
                     BN_, 4 * H_, H_, nullptr, nullptr, nullptr);
                lstm_point_k<<<dim3(32), dim3(256), 0, stream>>>(
                    zx + (long)ts * 4 * H_, (long)S_SRC * 4 * H_, zh,
                    syn, mem, m16, spkseq + (long)ts * H_, (long)S_SRC * H_,
                    F(ithr), BN_, H_);
            }
            gemm(spkseq, lin16, F(ilinb), linspk, H_, nullptr, 0,
                 BSs, H_, H_, nullptr, nullptr, nullptr);
            concat_seq_f16_k<<<dim3(1024), dim3(256), 0, stream>>>(
                linspk, x32, catbuf, BN_, S_SRC, H_, ind, 0);
            gemm(catbuf, mix16, F(imixb), o32, H_, o16, H_,
                 BSs, H_, H_ + ind, F(ina), F(ing), F(inb));
        }
        // decoder-init state projections
        concat2_rows_k<<<dim3(64), dim3(256), 0, stream>>>(synF, H_, H_, synB, H_, H_,
                                                           cat8, nullptr, BN_);
        gemm(cat8, synfc16, F(SYNFC_B), synD[li], H_, nullptr, 0,
             BN_, H_, 2 * H_, nullptr, nullptr, nullptr);
        concat2_rows_k<<<dim3(64), dim3(256), 0, stream>>>(memF, H_, H_, memB, H_, H_,
                                                           cat8, nullptr, BN_);
        gemm(cat8, memfc16, F(MEMFC_B), memD[li], H_, memD16[li], H_,
             BN_, H_, 2 * H_, nullptr, nullptr, nullptr);
        // encoder output (flip backward stream back)
        concat_seq_f16_k<<<dim3(1024), dim3(256), 0, stream>>>(
            nxtF32, nxtB32, catbuf, BN_, S_SRC, H_, H_, 1);
        gemm(catbuf, outfc16, F(OUTFC_B), encO32[li], H_, encO16[li], H_,
             BSs, H_, 2 * H_, nullptr, nullptr, nullptr);
        // swap cur <-> nxt
        { float* t0; _Float16* t1;
          t0 = curF32; curF32 = nxtF32; nxtF32 = t0;  t1 = curF16; curF16 = nxtF16; nxtF16 = t1;
          t0 = curB32; curB32 = nxtB32; nxtB32 = t0;  t1 = curB16; curB16 = nxtB16; nxtB16 = t1; }
    }

    // =================== decoder weights + k/v precompute ===================
    const _Float16 *qw16[2], *dwih16[2], *dwhh16[2], *dlin16[2], *dmix16[2];
    for (int i = 0; i < 2; ++i) {
        const int db = 6 + i * 21;               // attn: kW,kb,na,nb,ng,qW,qb,vW,vb
        const int sb = db + 9;                   // sl:   Whh,Wih,bhh,bih,linW,linb,mixW,mixb,na,nb,ng,thr
        const int ind = (i == 0) ? (E_ + H_) : 2 * H_;
        const _Float16* kw16 = castW(db + 0, (size_t)H_ * H_);
        qw16[i]              = castW(db + 5, (size_t)H_ * H_);
        const _Float16* vw16 = castW(db + 7, (size_t)H_ * H_);
        dwhh16[i] = castW(sb + 0, (size_t)4 * H_ * H_);
        dwih16[i] = castW(sb + 1, (size_t)4 * H_ * ind);
        dlin16[i] = castW(sb + 4, (size_t)H_ * H_);
        dmix16[i] = castW(sb + 6, (size_t)H_ * (H_ + ind));
        gemm(encO16[i], kw16, F(db + 1), k32[i], H_, nullptr, 0,
             BSs, H_, H_, nullptr, nullptr, nullptr);
        gemm(encO16[i], vw16, F(db + 8), v32[i], H_, nullptr, 0,
             BSs, H_, H_, nullptr, nullptr, nullptr);
    }

    // =================== decoder ===================
    embed_dyt_k<<<dim3(512), dim3(256), 0, stream>>>(tgt, F(DEC_EMB), F(DEC_NA),
                                                     F(DEC_NG), F(DEC_NB),
                                                     temb, nullptr, (long)BSt, E_);

    for (int ts = 0; ts < S_TGT; ++ts) {
        const float* prevA = temb + (long)ts * E_;
        long prevLd = (long)S_TGT * E_;
        int  prevK  = E_;
        for (int i = 0; i < 2; ++i) {
            const int db = 6 + i * 21;
            const int sb = db + 9;
            const int ind = H_ + prevK;
            // attention: q from current mem
            gemm(memD16[i], qw16[i], F(db + 6), q32, H_, nullptr, 0,
                 BN_, H_, H_, nullptr, nullptr, nullptr);
            attn_step_k<<<dim3(BN_), dim3(256), 0, stream>>>(
                q32, k32[i], v32[i], F(db + 2), F(db + 4), F(db + 3), ctx32, S_SRC, H_);
            // inp = concat(out, ctx)
            concat2_rows_k<<<dim3(64), dim3(256), 0, stream>>>(
                prevA, prevLd, prevK, ctx32, H_, H_, inp16, inp32, BN_);
            gemm(inp16, dwih16[i], F(sb + 3), zxd, 4 * H_, nullptr, 0,
                 BN_, 4 * H_, ind, nullptr, nullptr, nullptr);
            gemm(memD16[i], dwhh16[i], F(sb + 2), zh, 4 * H_, nullptr, 0,
                 BN_, 4 * H_, H_, nullptr, nullptr, nullptr);
            lstm_point_k<<<dim3(32), dim3(256), 0, stream>>>(
                zxd, (long)4 * H_, zh, synD[i], memD[i], memD16[i],
                spkd, (long)H_, F(sb + 11), BN_, H_);
            gemm(spkd, dlin16[i], F(sb + 5), linspkd, H_, nullptr, 0,
                 BN_, H_, H_, nullptr, nullptr, nullptr);
            concat2_rows_k<<<dim3(64), dim3(256), 0, stream>>>(
                linspkd, H_, H_, inp32, ind, ind, mixind, nullptr, BN_);
            if (i == 0) {
                gemm(mixind, dmix16[i], F(sb + 7), outL32, H_, outL16, H_,
                     BN_, H_, H_ + ind, F(sb + 8), F(sb + 10), F(sb + 9));
                prevA = outL32; prevLd = H_; prevK = H_;
            } else {
                gemm(mixind, dmix16[i], F(sb + 7),
                     dec32 + (long)ts * H_, S_TGT * H_,
                     dec16 + (long)ts * H_, S_TGT * H_,
                     BN_, H_, H_ + ind, F(sb + 8), F(sb + 10), F(sb + 9));
            }
        }
    }

    // =================== vocab projection ===================
    gemm(dec16, fc16, F(FC_B), (float*)d_out, V_, nullptr, 0,
         BSt, V_, H_, nullptr, nullptr, nullptr);
}
